// DynamicSparseAttention_57690000720284
// MI455X (gfx1250) — matrix operand order repaired and checked by reading, not compiled, not run on hardware
//
#include <hip/hip_runtime.h>
#include <math.h>
#include <stdint.h>

// Problem constants
#define BB   2
#define SS   2048
#define DD   1024
#define HH   16
#define KSEL 32
#define HD   64
#define NROW (BB*SS)          // 4096

typedef __attribute__((ext_vector_type(16))) __bf16 v16bf;
typedef __attribute__((ext_vector_type(8)))  __bf16 v8bf;
typedef __attribute__((ext_vector_type(8)))  float  v8f;

#if defined(__has_builtin)
#  if __has_builtin(__builtin_amdgcn_tensor_load_to_lds)
#    define HAVE_TDM 1
#  endif
#endif

static __device__ __forceinline__ v8f wmma_bf16(v16bf a, v16bf b, v8f c) {
    return __builtin_amdgcn_wmma_f32_16x16x32_bf16(false, a, false, b, (short)0, c, false, false);
}

// A-fragment loader for 16x16x32 bf16 WMMA.
// lane<16: row M=lane, k = {0..15}; lane>=16: row M=lane-16, k = {16..31}.
// p points at (row, k0) in a row-major bf16 matrix (global or LDS).
static __device__ __forceinline__ v16bf load_a16(const __bf16* p, int hi) {
    v16bf r = *(const v16bf*)(p + hi * 16);
    return r;
}

#ifdef HAVE_TDM
typedef __attribute__((ext_vector_type(4))) unsigned int u32x4;
typedef __attribute__((ext_vector_type(8))) int          i32x8;
typedef __attribute__((ext_vector_type(4))) int          i32x4;

// TDM: load a 32(x) x 128(y) bf16 tile (row stride = row_stride_elems) into LDS.
// D# layout per CDNA5 ISA (2D tile, groups 2/3 zero).
// This toolchain exposes the 6-arg builtin (g0, g1, g2, g3, g4, cpol).
static __device__ __forceinline__ void tdm_load_tile32x128(unsigned lds_byte_off,
                                                           const __bf16* gtile,
                                                           unsigned row_stride_elems) {
    unsigned long long ga = (unsigned long long)(size_t)gtile;   // byte address
    u32x4 g0;
    g0[0] = 1u;                                                  // count=1 valid descriptor
    g0[1] = lds_byte_off;                                        // lds_addr
    g0[2] = (unsigned)ga;                                        // global_addr[31:0]
    g0[3] = (unsigned)((ga >> 32) & 0x1FFFFFFu) | (2u << 30);    // addr[56:32] | type=2
    i32x8 g1;
    g1[0] = (int)(1u << 16);                                     // data_size=1 (2B), wg_mask=0
    g1[1] = (int)((row_stride_elems & 0xFFFFu) << 16);           // tensor_dim0[15:0]
    g1[2] = (int)(((row_stride_elems >> 16) & 0xFFFFu)           // tensor_dim0[31:16]
                  | (4096u << 16));                              // tensor_dim1[15:0]
    g1[3] = (int)(32u << 16);                                    // tensor_dim1 hi=0 | tile_dim0=32
    g1[4] = (int)128u;                                           // tile_dim1=128, tile_dim2=0
    g1[5] = (int)row_stride_elems;                               // tensor_dim0_stride[31:0]
    g1[6] = 0;                                                   // stride hi / dim1_stride lo
    g1[7] = 0;
    i32x4 z4 = {0, 0, 0, 0};
    i32x8 z8 = {0, 0, 0, 0, 0, 0, 0, 0};
    __builtin_amdgcn_tensor_load_to_lds(g0, g1, z4, z4, z8, 0);
}
#endif

// ---------------------------------------------------------------- convert
__global__ void cvt_bf16_kernel(const float* __restrict__ in, __bf16* __restrict__ out, int n) {
    for (int i = blockIdx.x * blockDim.x + threadIdx.x; i < n; i += gridDim.x * blockDim.x)
        out[i] = (__bf16)in[i];
}

// ---------------------------------------------------------------- importance: imp[n] = x[n,:]·ws + bs
__global__ void importance_kernel(const float* __restrict__ x, const float* __restrict__ ws,
                                  const float* __restrict__ bs, float* __restrict__ imp) {
    int wid  = blockIdx.x * (blockDim.x >> 5) + (threadIdx.x >> 5);
    int lane = threadIdx.x & 31;
    if (wid >= NROW) return;
    const float* row = x + (size_t)wid * DD;
    float acc = 0.f;
    for (int k = lane; k < DD; k += 32) acc += row[k] * ws[k];
#pragma unroll
    for (int m = 16; m >= 1; m >>= 1) acc += __shfl_xor(acc, m);
    if (lane == 0) imp[wid] = acc + bs[0];
}

// ---------------------------------------------------------------- top-32 per batch
__global__ void topk_kernel(const float* __restrict__ imp, int* __restrict__ topidx) {
    __shared__ float sv[SS];
    __shared__ float rmax[256];
    __shared__ int   ridx[256];
    int b = blockIdx.x, t = threadIdx.x;
    for (int i = t; i < SS; i += 256) sv[i] = imp[b * SS + i];
    __syncthreads();
    for (int it = 0; it < KSEL; ++it) {
        float m = -3.4e38f; int mi = SS;
        for (int i = t; i < SS; i += 256) {
            float v = sv[i];
            if (v > m || (v == m && i < mi)) { m = v; mi = i; }
        }
        rmax[t] = m; ridx[t] = mi;
        __syncthreads();
        for (int off = 128; off > 0; off >>= 1) {
            if (t < off) {
                if (rmax[t + off] > rmax[t] ||
                    (rmax[t + off] == rmax[t] && ridx[t + off] < ridx[t])) {
                    rmax[t] = rmax[t + off]; ridx[t] = ridx[t + off];
                }
            }
            __syncthreads();
        }
        if (t == 0) { topidx[b * KSEL + it] = ridx[0]; sv[ridx[0]] = -3.4e38f; }
        __syncthreads();
    }
}

// ---------------------------------------------------------------- gather Kg [B,H,32,64] and VgT [B,H,64,32]
__global__ void gather_kernel(const __bf16* __restrict__ Kh, const __bf16* __restrict__ Vh,
                              const int* __restrict__ topidx,
                              __bf16* __restrict__ Kg, __bf16* __restrict__ VgT) {
    int t = blockIdx.x * blockDim.x + threadIdx.x;   // B*H*32*64 = 65536
    int d = t & 63;
    int j = (t >> 6) & 31;
    int h = (t >> 11) & 15;
    int b = t >> 15;
    int s = topidx[b * KSEL + j];
    size_t src = (size_t)(b * SS + s) * DD + h * HD + d;
    int bh = b * HH + h;
    Kg[((size_t)bh * KSEL + j) * HD + d]  = Kh[src];
    VgT[((size_t)bh * HD + d) * KSEL + j] = Vh[src];
}

// ---------------------------------------------------------------- vsum[b,h,d] = sum_s V[b,s,h,d]
__global__ void vsum_kernel(const __bf16* __restrict__ Vh, float* __restrict__ vsum) {
    int bh = blockIdx.x;                  // B*H blocks, 64 threads
    int b = bh / HH, h = bh % HH;
    int d = threadIdx.x;
    const __bf16* p = Vh + (size_t)b * SS * DD + h * HD + d;
    float acc = 0.f;
    for (int s = 0; s < SS; ++s) acc += (float)p[(size_t)s * DD];
    vsum[bh * HD + d] = acc;
}

// ---------------------------------------------------------------- tiled GEMM: C[M,N] = A[M,K]·W[N,K]^T + bias
// 128x128 block tile, 8 waves, each wave 32x64 (8 WMMAs / 32-wide K chunk).
// K chunks staged via TDM tensor_load_to_lds (double-buffered) with s_wait_tensorcnt.
template <bool OUTF32>
__global__ void __launch_bounds__(256)
gemm_tiled_kernel(const __bf16* __restrict__ A, const __bf16* __restrict__ W,
                  const float* __restrict__ bias, void* __restrict__ Cout,
                  int M, int N, int Kd) {
    __shared__ __align__(16) __bf16 smem[2][2][128 * 32];    // [buf][A|B][row*32+k] = 32 KB
    const int ntn  = N >> 7;
    const int tm   = blockIdx.x / ntn, tn = blockIdx.x % ntn;
    const int tid  = threadIdx.x;
    const int wid  = tid >> 5, lane = tid & 31, hi = lane >> 4, ln = lane & 15;
    const int wm   = wid >> 1, wn = wid & 1;                 // 4x2 wave grid

    const __bf16* Abase = A + (size_t)tm * 128 * Kd;
    const __bf16* Wbase = W + (size_t)tn * 128 * Kd;
    const int nk = Kd >> 5;

    v8f acc[2][4];
#pragma unroll
    for (int mi = 0; mi < 2; ++mi)
#pragma unroll
        for (int sub = 0; sub < 4; ++sub) acc[mi][sub] = v8f{};

    auto stage = [&](int k0, int p) {
#ifdef HAVE_TDM
        if (wid == 0) {
            tdm_load_tile32x128((unsigned)(p * 16384),        Abase + k0, (unsigned)Kd);
            tdm_load_tile32x128((unsigned)(p * 16384 + 8192), Wbase + k0, (unsigned)Kd);
        }
#else
        int row = tid >> 1, half = (tid & 1) << 4;
        *(v16bf*)(&smem[p][0][row * 32 + half]) = *(const v16bf*)(Abase + (size_t)row * Kd + k0 + half);
        *(v16bf*)(&smem[p][1][row * 32 + half]) = *(const v16bf*)(Wbase + (size_t)row * Kd + k0 + half);
#endif
    };

    stage(0, 0);
    for (int i = 0; i < nk; ++i) {
        const int p = i & 1;
        if (i + 1 < nk) stage((i + 1) << 5, p ^ 1);
#ifdef HAVE_TDM
        if (wid == 0) {
            if (i + 1 < nk) __builtin_amdgcn_s_wait_tensorcnt(2);
            else            __builtin_amdgcn_s_wait_tensorcnt(0);
        }
#endif
        __syncthreads();
        const __bf16* At = &smem[p][0][0];
        const __bf16* Bt = &smem[p][1][0];
        v16bf a[2];
#pragma unroll
        for (int mi = 0; mi < 2; ++mi)
            a[mi] = load_a16(At + (wm * 32 + mi * 16 + ln) * 32, hi);
#pragma unroll
        for (int sub = 0; sub < 4; ++sub) {
            v16bf b = *(const v16bf*)(Bt + (wn * 64 + sub * 16 + ln) * 32 + hi * 16);
            acc[0][sub] = wmma_bf16(a[0], b, acc[0][sub]);
            acc[1][sub] = wmma_bf16(a[1], b, acc[1][sub]);
        }
        __syncthreads();
    }

#pragma unroll
    for (int mi = 0; mi < 2; ++mi)
#pragma unroll
        for (int sub = 0; sub < 4; ++sub) {
            int col = tn * 128 + wn * 64 + sub * 16 + ln;
            float bv = bias[col];
#pragma unroll
            for (int r = 0; r < 8; ++r) {
                int row = tm * 128 + wm * 32 + mi * 16 + hi * 8 + r;
                float v = acc[mi][sub][r] + bv;
                if (OUTF32) ((float*)Cout)[(size_t)row * N + col] = v;
                else        ((__bf16*)Cout)[(size_t)row * N + col] = (__bf16)v;
            }
        }
}

// ---------------------------------------------------------------- attention (one wave = 16 queries of one (b,h))
__global__ void attn_kernel(const __bf16* __restrict__ Q, const __bf16* __restrict__ Kg,
                            const __bf16* __restrict__ VgT, const float* __restrict__ vsum,
                            __bf16* __restrict__ Aout) {
    __shared__ __align__(32) __bf16 plds[4][16][32];
    int w    = threadIdx.x >> 5;
    int wid  = blockIdx.x * 4 + w;          // 0..4095
    int qt   = wid & 127;
    int h    = (wid >> 7) & 15;
    int b    = wid >> 11;
    int lane = threadIdx.x & 31, hi = lane >> 4, ln = lane & 15;
    int bh   = b * HH + h;

    // ---- scores: Q(16x64) x Kg^T -> 16x32 (two 16x16 tiles)
    const __bf16* qbase = Q + (size_t)(b * SS + qt * 16 + ln) * DD + h * HD;
    const __bf16* kb    = Kg + (size_t)bh * KSEL * HD;
    v8f s0 = {}, s1 = {};
#pragma unroll
    for (int k0 = 0; k0 < HD; k0 += 32) {
        v16bf a  = load_a16(qbase + k0, hi);
        v16bf b0 = *(const v16bf*)(kb + (size_t)ln * HD + k0 + hi * 16);        // keys 0..15
        v16bf b1 = *(const v16bf*)(kb + (size_t)(16 + ln) * HD + k0 + hi * 16); // keys 16..31
        s0 = wmma_bf16(a, b0, s0);
        s1 = wmma_bf16(a, b1, s1);
    }

    // ---- softmax algebra: denom = sum_topk e^s + (S-K); weights p = e^s - 1
    const float scale = 0.125f;              // 1/sqrt(64)
    float dinv[8], p0s[8], p1s[8];
#pragma unroll
    for (int r = 0; r < 8; ++r) {
        float e0 = expf(s0[r] * scale);
        float e1 = expf(s1[r] * scale);
        float ps = e0 + e1;
        ps += __shfl_xor(ps, 1);
        ps += __shfl_xor(ps, 2);
        ps += __shfl_xor(ps, 4);
        ps += __shfl_xor(ps, 8);             // sum over 16 lanes of this half => 32 keys
        dinv[r] = 1.0f / (ps + (float)(SS - KSEL));
        p0s[r] = e0 - 1.0f;
        p1s[r] = e1 - 1.0f;
    }

    // ---- transpose p (C layout: lane=key, vgpr=query) into A layout via LDS
#pragma unroll
    for (int r = 0; r < 8; ++r) {
        plds[w][hi * 8 + r][ln]      = (__bf16)p0s[r];
        plds[w][hi * 8 + r][16 + ln] = (__bf16)p1s[r];
    }
    __syncthreads();
    v16bf pa = *(const v16bf*)(&plds[w][ln][hi * 16]);

    // ---- out = P(16x32) x Vg(32x64) ; B fragments from transposed VgT[d][j]
    const __bf16* vb = VgT + (size_t)bh * HD * KSEL;
    v8f o[4] = {v8f{}, v8f{}, v8f{}, v8f{}};
#pragma unroll
    for (int sub = 0; sub < 4; ++sub) {
        v16bf bf = *(const v16bf*)(vb + (size_t)(sub * 16 + ln) * KSEL + hi * 16);
        o[sub] = wmma_bf16(pa, bf, o[sub]);
    }

    // ---- add V column-sum, scale by 1/denom, scatter back to [B,S,D]
#pragma unroll
    for (int sub = 0; sub < 4; ++sub) {
        float vs = vsum[bh * HD + sub * 16 + ln];
#pragma unroll
        for (int r = 0; r < 8; ++r) {
            float val = (o[sub][r] + vs) * dinv[r];
            Aout[(size_t)(b * SS + qt * 16 + hi * 8 + r) * DD + h * HD + sub * 16 + ln] = (__bf16)val;
        }
    }
}

// ---------------------------------------------------------------- launch
extern "C" void kernel_launch(void* const* d_in, const int* in_sizes, int n_in,
                              void* d_out, int out_size, void* d_ws, size_t ws_size,
                              hipStream_t stream) {
    const float* x  = (const float*)d_in[0];
    const float* wq = (const float*)d_in[1];
    const float* bq = (const float*)d_in[2];
    const float* wk = (const float*)d_in[3];
    const float* bk = (const float*)d_in[4];
    const float* wv = (const float*)d_in[5];
    const float* bv = (const float*)d_in[6];
    const float* wo = (const float*)d_in[7];
    const float* bo = (const float*)d_in[8];
    const float* ws = (const float*)d_in[9];
    const float* bs = (const float*)d_in[10];

    char* base = (char*)d_ws;
    __bf16* xh    = (__bf16*)(base + 0);                      // 8 MiB
    __bf16* wqh   = (__bf16*)(base + (8u << 20));             // 2 MiB each
    __bf16* wkh   = (__bf16*)(base + (10u << 20));
    __bf16* wvh   = (__bf16*)(base + (12u << 20));
    __bf16* woh   = (__bf16*)(base + (14u << 20));
    __bf16* Qh    = (__bf16*)(base + (16u << 20));            // 8 MiB
    __bf16* Kh    = (__bf16*)(base + (24u << 20));
    __bf16* Vh    = (__bf16*)(base + (32u << 20));
    __bf16* Ah    = (__bf16*)(base + (40u << 20));            // attention output
    float*  imp   = (float*)(base + (48u << 20));             // 16 KiB
    int*    tidx  = (int*)(base + (48u << 20) + (32u << 10)); // 256 B
    __bf16* Kg    = (__bf16*)(base + (48u << 20) + (64u << 10));   // 128 KiB
    __bf16* VgT   = (__bf16*)(base + (48u << 20) + (192u << 10));  // 128 KiB
    float*  vsum  = (float*)(base + (48u << 20) + (320u << 10));   // 8 KiB

    // 1. fp32 -> bf16
    cvt_bf16_kernel<<<1024, 256, 0, stream>>>(x,  xh,  NROW * DD);
    cvt_bf16_kernel<<<512,  256, 0, stream>>>(wq, wqh, DD * DD);
    cvt_bf16_kernel<<<512,  256, 0, stream>>>(wk, wkh, DD * DD);
    cvt_bf16_kernel<<<512,  256, 0, stream>>>(wv, wvh, DD * DD);
    cvt_bf16_kernel<<<512,  256, 0, stream>>>(wo, woh, DD * DD);

    // 2. token importance + 3. top-k
    importance_kernel<<<NROW / 8, 256, 0, stream>>>(x, ws, bs, imp);
    topk_kernel<<<BB, 256, 0, stream>>>(imp, tidx);

    // 5. Q/K/V projections: 32x8 = 256 blocks of 128x128 tiles
    gemm_tiled_kernel<false><<<256, 256, 0, stream>>>(xh, wqh, bq, Qh, NROW, DD, DD);
    gemm_tiled_kernel<false><<<256, 256, 0, stream>>>(xh, wkh, bk, Kh, NROW, DD, DD);
    gemm_tiled_kernel<false><<<256, 256, 0, stream>>>(xh, wvh, bv, Vh, NROW, DD, DD);

    // 4. gather top-K keys/values + V column sums
    gather_kernel<<<256, 256, 0, stream>>>(Kh, Vh, tidx, Kg, VgT);
    vsum_kernel<<<BB * HH, 64, 0, stream>>>(Vh, vsum);

    // 6. sparse attention (4096 waves)
    attn_kernel<<<1024, 128, 0, stream>>>(Qh, Kg, VgT, vsum, Ah);

    // 7. output projection -> fp32 d_out
    gemm_tiled_kernel<true><<<256, 256, 0, stream>>>(Ah, woh, bo, (float*)d_out, NROW, DD, DD);
}
